// Variacne_Block_18889266167836
// MI455X (gfx1250) — matrix operand order repaired and checked by reading, not compiled, not run on hardware
//
#include <hip/hip_runtime.h>
#include <hip/hip_bf16.h>

typedef __attribute__((ext_vector_type(16))) _Float16 v16h;
typedef __attribute__((ext_vector_type(8)))  float    v8f;

#define NB_B    8
#define NC      512
#define T_ENC   512
#define T_PR    256
#define T_DEC   2048
#define N_STACK 3
#define N_CONVB 3
#define N_HEADS 8
#define D_HEAD  64
#define EPS     1e-5f

// ---------------------------------------------------------------------------
// Fragment packing (f32 -> f16 in exact gfx1250 WMMA fragment order)
//
// A fragment (16x32, MxK), tile index (mt,kt), tile = 512 halves:
//   half slot (lane la, elem el):  m = mt*16 + (la&15)
//                                  k = kt*32 + (el&7) + ((la>>4)<<3) + ((el>>3)<<4)
// B fragment (32x16, KxN):         n = nt*16 + (la&15)
//                                  k = kt*32 + (el&7) + ((la>>4)<<3) + ((el>>3)<<4)
// A pack layout: [z][mt][kt][512]   B pack layout: [z][nt][kt][512]
// => in the GEMM, each lane's v16h operand is one contiguous 32B global load.
// ---------------------------------------------------------------------------
__global__ __launch_bounds__(256)
void k_pack_a(const float* __restrict__ in, _Float16* __restrict__ outp,
              int M, int K, int lda, int ae, long long sZ, int Kt,
              long long total)
{
    long long i = (long long)blockIdx.x * 256 + threadIdx.x;
    if (i >= total) return;
    int w    = (int)(i & 511);
    long long tile = i >> 9;
    int kt = (int)(tile % Kt);
    long long zm = tile / Kt;                  // z*Mt + mt
    int Mt = (M + 15) >> 4;
    int mt = (int)(zm % Mt);
    int z  = (int)(zm / Mt);
    int la = w >> 4, el = w & 15;
    int m = mt * 16 + (la & 15);
    int k = kt * 32 + (el & 7) + ((la >> 4) << 3) + ((el >> 3) << 4);
    float v = 0.f;
    if (m < M && k < K)
        v = in[(long long)z * sZ + (long long)m * lda + (long long)k * ae];
    outp[i] = (_Float16)v;
}

__global__ __launch_bounds__(256)
void k_pack_b(const float* __restrict__ in, _Float16* __restrict__ outp,
              int K, int N, int ldb, int bcs, long long sZ, int Kt,
              int conv3, int Tsrc, long long total)
{
    long long i = (long long)blockIdx.x * 256 + threadIdx.x;
    if (i >= total) return;
    int w = (int)(i & 511);
    long long tile = i >> 9;
    int kt = (int)(tile % Kt);
    long long zn = tile / Kt;                  // z*Nt + nt
    int Nt = (N + 15) >> 4;
    int nt = (int)(zn % Nt);
    int z  = (int)(zn / Nt);
    int la = w >> 4, el = w & 15;
    int n = nt * 16 + (la & 15);
    int k = kt * 32 + (el & 7) + ((la >> 4) << 3) + ((el >> 3) << 4);
    float v = 0.f;
    if (k < K && n < N) {
        if (conv3) {
            int ci = k / 3, p = k - ci * 3;    // im2col row (i,tap), col shift
            int t = n + p - 1;
            if (t >= 0 && t < Tsrc)
                v = in[(long long)z * sZ + (long long)ci * ldb + t];
        } else {
            v = in[(long long)z * sZ + (long long)k * ldb + (long long)n * bcs];
        }
    }
    outp[i] = (_Float16)v;
}

// ---------------------------------------------------------------------------
// WMMA GEMM on pre-packed fragments.  No LDS, no predication in the K loop.
// Block = 128 threads (4 waves).  Each wave owns MF M-fragments (16 rows each)
// that SHARE the block's four B-fragments -> MF*4 independent WMMAs per K-step.
// MF=2: tile 128x64 (big GEMMs, M % 128 == 0)
// MF=1: tile  64x64 (per-head GEMMs, M % 64 == 0)
// ---------------------------------------------------------------------------
template <int MF>
__global__ __launch_bounds__(128)
void wmma_gemm(const _Float16* __restrict__ Ap, const _Float16* __restrict__ Bp,
               const float* __restrict__ bias, const float* __restrict__ Rs,
               float* __restrict__ D,
               int M, int N, int Kt, int ldd,
               long long sAz, long long sBz, long long sDz)
{
    const int z    = blockIdx.z;
    const int m0   = blockIdx.y * (64 * MF);
    const int n0   = blockIdx.x * 64;
    const int wave = threadIdx.x >> 5;
    const int lane = threadIdx.x & 31;

    const _Float16* Abase = Ap + z * sAz
                          + ((long long)((m0 >> 4) + wave * MF) * Kt) * 512 + lane * 16;
    const _Float16* Bbase = Bp + z * sBz
                          + ((long long)(n0 >> 4) * Kt) * 512 + lane * 16;

    v8f acc[MF][4] = {};
    for (int kt = 0; kt < Kt; ++kt) {
        int pk = (kt + 4 < Kt) ? (kt + 4) : (Kt - 1);   // clamped deep-K prefetch
        __builtin_prefetch(Abase + (long long)pk * 512, 0, 1);
        __builtin_prefetch(Bbase + (long long)pk * 512, 0, 1);

        v16h a[MF];
        #pragma unroll
        for (int mf = 0; mf < MF; ++mf)
            a[mf] = *reinterpret_cast<const v16h*>(
                Abase + (long long)(mf * Kt + kt) * 512);
        #pragma unroll
        for (int f = 0; f < 4; ++f) {
            v16h b = *reinterpret_cast<const v16h*>(
                Bbase + (long long)(f * Kt + kt) * 512);
            #pragma unroll
            for (int mf = 0; mf < MF; ++mf)
                acc[mf][f] = __builtin_amdgcn_wmma_f32_16x16x32_f16(
                    false, a[mf], false, b, (short)0, acc[mf][f], false, false);
        }
    }

    // epilogue: C/D 16x16 f32 layout — VGPR r: M=r (lanes 0-15), M=r+8 (16-31)
    const int nl = lane & 15;
    const int mh = (lane >> 4) << 3;
    float* Db = D + z * sDz;
    const float* Rb = Rs ? (Rs + z * sDz) : nullptr;
    #pragma unroll
    for (int mf = 0; mf < MF; ++mf) {
        #pragma unroll
        for (int f = 0; f < 4; ++f) {
            int gn = n0 + f * 16 + nl;
            if (gn >= N) continue;
            #pragma unroll
            for (int r = 0; r < 8; ++r) {
                int gm = m0 + (wave * MF + mf) * 16 + mh + r;
                if (gm >= M) continue;
                float v = acc[mf][f][r];
                if (bias) v += bias[gm];
                long long off = (long long)gm * ldd + gn;
                if (Rb) v += Rb[off];
                Db[off] = v;
            }
        }
    }
}

// ---------------------------------------------------------------------------
// elementwise / reduction kernels
// ---------------------------------------------------------------------------
__global__ __launch_bounds__(256)
void k_rms_silu_res(const float* __restrict__ Y, const float* __restrict__ g,
                    float* __restrict__ X, int T)
{
    int bt = blockIdx.x;
    int b = bt / T, t = bt - b * T;
    const float* yc = Y + (long long)b * NC * T + t;
    float*       xc = X + (long long)b * NC * T + t;
    __shared__ float red[256];
    float s = 0.f;
    for (int c = threadIdx.x; c < NC; c += 256) {
        float v = yc[(long long)c * T];
        s += v * v;
    }
    red[threadIdx.x] = s;
    __syncthreads();
    for (int st = 128; st > 0; st >>= 1) {
        if (threadIdx.x < st) red[threadIdx.x] += red[threadIdx.x + st];
        __syncthreads();
    }
    float rinv = rsqrtf(red[0] / (float)NC + EPS);
    for (int c = threadIdx.x; c < NC; c += 256) {
        float v = yc[(long long)c * T] * rinv * g[c];
        float sl = v / (1.f + __expf(-v));
        xc[(long long)c * T] += sl;
    }
}

__global__ __launch_bounds__(64)
void k_softmax_q(float* __restrict__ q, int T)
{
    int gid = blockIdx.x;
    int t  = gid % T;
    int bh = gid / T;
    int h  = bh % N_HEADS, b = bh / N_HEADS;
    float* base = q + ((long long)b * NC + h * D_HEAD) * T + t;
    int d = threadIdx.x;
    float v = base[(long long)d * T];
    __shared__ float red[64];
    red[d] = v; __syncthreads();
    for (int s = 32; s > 0; s >>= 1) { if (d < s) red[d] = fmaxf(red[d], red[d + s]); __syncthreads(); }
    float mx = red[0]; __syncthreads();
    float e = __expf(v - mx);
    red[d] = e; __syncthreads();
    for (int s = 32; s > 0; s >>= 1) { if (d < s) red[d] += red[d + s]; __syncthreads(); }
    base[(long long)d * T] = e / red[0];
}

__global__ __launch_bounds__(256)
void k_softmax_k(float* __restrict__ k)
{
    float* base = k + (long long)blockIdx.x * T_PR;
    int t = threadIdx.x;
    float v = base[t];
    __shared__ float red[256];
    red[t] = v; __syncthreads();
    for (int s = 128; s > 0; s >>= 1) { if (t < s) red[t] = fmaxf(red[t], red[t + s]); __syncthreads(); }
    float mx = red[0]; __syncthreads();
    float e = __expf(v - mx);
    red[t] = e; __syncthreads();
    for (int s = 128; s > 0; s >>= 1) { if (t < s) red[t] += red[t + s]; __syncthreads(); }
    base[t] = e / red[0];
}

__global__ __launch_bounds__(256)
void k_proj(const float* __restrict__ x, const float* __restrict__ w,
            const float* __restrict__ pb, float* __restrict__ out,
            int T, int do_softplus, int total)
{
    int i = blockIdx.x * 256 + threadIdx.x;
    if (i >= total) return;
    int b = i / T, t = i - b * T;
    const float* xb = x + (long long)b * NC * T;
    float acc = pb[0];
    for (int c = 0; c < NC; ++c) {
        const float* row = xb + (long long)c * T + t;
        float xm = (t > 0)     ? row[-1] : 0.f;
        float x0 = row[0];
        float xp = (t < T - 1) ? row[1]  : 0.f;
        acc += w[c * 3] * xm + w[c * 3 + 1] * x0 + w[c * 3 + 2] * xp;
    }
    if (do_softplus) acc = (acc > 20.f) ? acc : log1pf(__expf(acc));
    out[i] = acc;
}

__global__ __launch_bounds__(256)
void k_frame_index(const int* __restrict__ dur, int* __restrict__ idx)
{
    __shared__ int cs[T_ENC + 1];
    int b = blockIdx.x;
    if (threadIdx.x == 0) {
        int a = 0; cs[0] = 0;
        for (int t = 0; t < T_ENC; ++t) { a += dur[b * T_ENC + t]; cs[t + 1] = a; }
    }
    __syncthreads();
    for (int d = threadIdx.x; d < T_DEC; d += 256) {
        int lo = 0, hi = T_ENC;
        while (hi - lo > 1) { int mid = (lo + hi) >> 1; if (cs[mid] <= d) lo = mid; else hi = mid; }
        idx[b * T_DEC + d] = lo;
    }
}

__global__ __launch_bounds__(256)
void k_expand(const float* __restrict__ enc, const int* __restrict__ idx,
              float* __restrict__ ex)
{
    long long i = (long long)blockIdx.x * 256 + threadIdx.x;
    if (i >= (long long)NB_B * NC * T_DEC) return;
    int d = (int)(i % T_DEC);
    long long bc = i / T_DEC;
    int b = (int)(bc / NC), c = (int)(bc % NC);
    ex[i] = enc[((long long)b * NC + c) * T_ENC + idx[b * T_DEC + d]];
}

__global__ __launch_bounds__(256)
void k_final(const float* __restrict__ ex, const float* __restrict__ f0s,
             const float* __restrict__ w, const float* __restrict__ bb,
             float* __restrict__ out)
{
    long long i = (long long)blockIdx.x * 256 + threadIdx.x;
    if (i >= (long long)NB_B * NC * T_DEC) return;
    int d = (int)(i % T_DEC);
    long long bc = i / T_DEC;
    int b = (int)(bc / NC), c = (int)(bc % NC);
    out[i] = ex[i] + w[c] * f0s[(long long)b * T_DEC + d] + bb[c];
}

// ---------------------------------------------------------------------------
// host-side helpers
// ---------------------------------------------------------------------------
static inline void pack_a(hipStream_t st, const float* in, _Float16* outp,
                          int M, int K, int lda, int ae, long long sZ, int Z)
{
    int Mt = (M + 15) >> 4, Kt = (K + 31) >> 5;
    long long total = (long long)Z * Mt * Kt * 512;
    k_pack_a<<<(unsigned)((total + 255) / 256), 256, 0, st>>>(
        in, outp, M, K, lda, ae, sZ, Kt, total);
}

static inline void pack_b(hipStream_t st, const float* in, _Float16* outp,
                          int K, int N, int ldb, int bcs, long long sZ, int Z,
                          int conv3, int Tsrc)
{
    int Nt = (N + 15) >> 4, Kt = (K + 31) >> 5;
    long long total = (long long)Z * Nt * Kt * 512;
    k_pack_b<<<(unsigned)((total + 255) / 256), 256, 0, st>>>(
        in, outp, K, N, ldb, bcs, sZ, Kt, conv3, Tsrc, total);
}

static inline void gemm(hipStream_t st, const _Float16* Ap, const _Float16* Bp,
                        const float* bias, const float* res, float* D,
                        int M, int N, int K, int ldd,
                        long long sAz, long long sBz, long long sDz, int Z)
{
    int Kt = (K + 31) >> 5;
    if ((M & 127) == 0) {           // big GEMMs: 128x64 tile, B-frag reuse x2
        dim3 g((N + 63) / 64, M / 128, Z);
        wmma_gemm<2><<<g, 128, 0, st>>>(Ap, Bp, bias, res, D, M, N, Kt,
                                        ldd, sAz, sBz, sDz);
    } else {                        // per-head GEMMs: 64x64 tile
        dim3 g((N + 63) / 64, (M + 63) / 64, Z);
        wmma_gemm<1><<<g, 128, 0, st>>>(Ap, Bp, bias, res, D, M, N, Kt,
                                        ldd, sAz, sBz, sDz);
    }
}

struct VpParams {
    const float *conv_w, *conv_b, *rms_g, *q_w, *q_b, *k_w, *k_b,
                *v_w, *v_b, *o_w, *o_b, *proj_w, *proj_b;
};
struct VpPacked {
    const _Float16* conv[N_STACK * N_CONVB];
    const _Float16 *q[N_STACK], *k[N_STACK], *v[N_STACK], *o[N_STACK];
};

#define CONV_PACK ((long long)32 * 48 * 512)   // M=512 (32 tiles) x K=1536 (48 tiles)
#define PROJ_PACK ((long long)32 * 16 * 512)   // M=512 x K=512

static void pack_vp_weights(hipStream_t st, const VpParams& p, _Float16*& cur,
                            VpPacked& o)
{
    for (int i = 0; i < N_STACK * N_CONVB; ++i) {
        pack_a(st, p.conv_w + (long long)i * NC * NC * 3, cur, NC, NC * 3, NC * 3, 1, 0, 1);
        o.conv[i] = cur; cur += CONV_PACK;
    }
    for (int s = 0; s < N_STACK; ++s) {
        pack_a(st, p.q_w + (long long)s * NC * NC, cur, NC, NC, NC, 1, 0, 1); o.q[s] = cur; cur += PROJ_PACK;
        pack_a(st, p.k_w + (long long)s * NC * NC, cur, NC, NC, NC, 1, 0, 1); o.k[s] = cur; cur += PROJ_PACK;
        pack_a(st, p.v_w + (long long)s * NC * NC, cur, NC, NC, NC, 1, 0, 1); o.v[s] = cur; cur += PROJ_PACK;
        pack_a(st, p.o_w + (long long)s * NC * NC, cur, NC, NC, NC, 1, 0, 1); o.o[s] = cur; cur += PROJ_PACK;
    }
}

static void run_vp(hipStream_t st, const VpParams& p, const VpPacked& pw,
                   float* X, const _Float16* PPR,
                   float* Y, float* Q, float* Kb, float* Vb, float* Ctx,
                   _Float16* PA, _Float16* PB,
                   int T, float* proj_out, int do_softplus)
{
    const long long sX  = (long long)NC * T;
    const long long sPr = (long long)NC * T_PR;
    const int Nt = T >> 4;
    for (int s = 0; s < N_STACK; ++s) {
        for (int cb = 0; cb < N_CONVB; ++cb) {
            // Y = conv(X) + b ; X += silu(rms(Y)*g)
            pack_b(st, X, PB, NC * 3, T, T, 1, sX, NB_B, 1, T);
            gemm(st, pw.conv[s * N_CONVB + cb], PB,
                 p.conv_b + (s * N_CONVB + cb) * NC, nullptr, Y,
                 NC, T, NC * 3, T, 0, (long long)Nt * 48 * 512, sX, NB_B);
            k_rms_silu_res<<<NB_B * T, 256, 0, st>>>(
                Y, p.rms_g + (s * N_CONVB + cb) * NC, X, T);
        }
        // --- linear attention ---
        pack_b(st, X, PB, NC, T, T, 1, sX, NB_B, 0, 0);
        gemm(st, pw.q[s], PB, p.q_b + s * NC, nullptr, Q,
             NC, T, NC, T, 0, (long long)Nt * 16 * 512, sX, NB_B);
        gemm(st, pw.k[s], PPR, p.k_b + s * NC, nullptr, Kb,
             NC, T_PR, NC, T_PR, 0, (long long)16 * 16 * 512, sPr, NB_B);
        gemm(st, pw.v[s], PPR, p.v_b + s * NC, nullptr, Vb,
             NC, T_PR, NC, T_PR, 0, (long long)16 * 16 * 512, sPr, NB_B);
        k_softmax_q<<<NB_B * N_HEADS * T, 64, 0, st>>>(Q, T);
        k_softmax_k<<<NB_B * NC, 256, 0, st>>>(Kb);
        // ctx[d,e] = sum_k kh[d,k]*vh[e,k]   per z=(b,h): A=kh, B=vh^T
        pack_a(st, Kb, PA, D_HEAD, T_PR, T_PR, 1, (long long)D_HEAD * T_PR, NB_B * N_HEADS);
        pack_b(st, Vb, PB, T_PR, D_HEAD, 1, T_PR, (long long)D_HEAD * T_PR, NB_B * N_HEADS, 0, 0);
        gemm(st, PA, PB, nullptr, nullptr, Ctx,
             D_HEAD, D_HEAD, T_PR, D_HEAD,
             (long long)4 * 8 * 512, (long long)4 * 8 * 512,
             (long long)D_HEAD * D_HEAD, NB_B * N_HEADS);
        // o[e,q] = sum_d ctx[d,e]*qh[d,q]   per z: A=ctx^T, B=qh   -> Y
        pack_a(st, Ctx, PA, D_HEAD, D_HEAD, 1, D_HEAD, (long long)D_HEAD * D_HEAD, NB_B * N_HEADS);
        pack_b(st, Q, PB, D_HEAD, T, T, 1, (long long)D_HEAD * T, NB_B * N_HEADS, 0, 0);
        gemm(st, PA, PB, nullptr, nullptr, Y,
             D_HEAD, T, D_HEAD, T,
             (long long)4 * 2 * 512, (long long)Nt * 2 * 512,
             (long long)D_HEAD * T, NB_B * N_HEADS);
        // X = o_w @ Y + o_b + X
        pack_b(st, Y, PB, NC, T, T, 1, sX, NB_B, 0, 0);
        gemm(st, pw.o[s], PB, p.o_b + s * NC, X, X,
             NC, T, NC, T, 0, (long long)Nt * 16 * 512, sX, NB_B);
    }
    int tot = NB_B * T;
    k_proj<<<(tot + 255) / 256, 256, 0, st>>>(X, p.proj_w, p.proj_b, proj_out,
                                              T, do_softplus, tot);
}

static VpParams make_params(void* const* d_in, int base)
{
    VpParams p;
    p.conv_w = (const float*)d_in[base + 0];
    p.conv_b = (const float*)d_in[base + 1];
    p.rms_g  = (const float*)d_in[base + 2];
    p.q_w    = (const float*)d_in[base + 3];
    p.q_b    = (const float*)d_in[base + 4];
    p.k_w    = (const float*)d_in[base + 5];
    p.k_b    = (const float*)d_in[base + 6];
    p.v_w    = (const float*)d_in[base + 7];
    p.v_b    = (const float*)d_in[base + 8];
    p.o_w    = (const float*)d_in[base + 9];
    p.o_b    = (const float*)d_in[base + 10];
    p.proj_w = (const float*)d_in[base + 11];
    p.proj_b = (const float*)d_in[base + 12];
    return p;
}

extern "C" void kernel_launch(void* const* d_in, const int* in_sizes, int n_in,
                              void* d_out, int out_size, void* d_ws, size_t ws_size,
                              hipStream_t stream)
{
    (void)in_sizes; (void)n_in; (void)out_size; (void)ws_size;
    const float* encodings = (const float*)d_in[0];
    const float* prompts   = (const float*)d_in[2];
    const int*   durations = (const int*)d_in[3];
    const float* f0s       = (const float*)d_in[4];
    VpParams dur_p = make_params(d_in, 6);
    VpParams f0_p  = make_params(d_in, 19);
    const float* f0_emb_w = (const float*)d_in[32];
    const float* f0_emb_b = (const float*)d_in[33];

    float* out_main = (float*)d_out;
    float* out_dur  = out_main + (long long)NB_B * NC * T_DEC;
    float* out_f0   = out_dur + NB_B * T_ENC;

    // ---- workspace carve (256B aligned chunks) ----
    char* ws = (char*)d_ws;
    size_t off = 0;
    auto carve_f = [&](long long nfl) {
        float* p = (float*)(ws + off);
        off = (off + (size_t)nfl * 4 + 255) & ~(size_t)255;
        return p;
    };
    auto carve_h = [&](long long nh) {
        _Float16* p = (_Float16*)(ws + off);
        off = (off + (size_t)nh * 2 + 255) & ~(size_t)255;
        return p;
    };
    const long long NBIG = (long long)NB_B * NC * T_DEC;
    float* EXP = carve_f(NBIG);
    float* X   = carve_f(NBIG);
    float* Y   = carve_f(NBIG);
    float* Q   = carve_f(NBIG);
    float* Kb  = carve_f((long long)NB_B * NC * T_PR);
    float* Vb  = carve_f((long long)NB_B * NC * T_PR);
    float* Ctx = carve_f((long long)NB_B * N_HEADS * D_HEAD * D_HEAD);
    int*   IDX = (int*)(ws + off);
    off = (off + (size_t)NB_B * T_DEC * 4 + 255) & ~(size_t)255;
    // packed-weight pool: 18 conv + 24 proj matrices
    _Float16* PW = carve_h(18 * CONV_PACK + 24 * PROJ_PACK);
    _Float16* PPR = carve_h((long long)NB_B * 16 * 16 * 512);       // packed prompts
    _Float16* PA  = carve_h((long long)NB_B * N_HEADS * 4 * 8 * 512);
    _Float16* PB  = carve_h((long long)NB_B * 48 * (T_DEC >> 4) * 512);

    // ---- pack all launch-constant operands once ----
    _Float16* cur = PW;
    VpPacked dur_pw, f0_pw;
    pack_vp_weights(stream, dur_p, cur, dur_pw);
    pack_vp_weights(stream, f0_p,  cur, f0_pw);
    pack_b(stream, prompts, PPR, NC, T_PR, T_PR, 1, (long long)NC * T_PR, NB_B, 0, 0);

    // ---- phase 1: duration predictor on encodings (T=512) ----
    hipMemcpyAsync(X, encodings, (size_t)NB_B * NC * T_ENC * 4,
                   hipMemcpyDeviceToDevice, stream);
    run_vp(stream, dur_p, dur_pw, X, PPR, Y, Q, Kb, Vb, Ctx, PA, PB,
           T_ENC, out_dur, 1);

    // ---- phase 2: length regulate (cumsum + gather) ----
    k_frame_index<<<NB_B, 256, 0, stream>>>(durations, IDX);
    k_expand<<<(unsigned)((NBIG + 255) / 256), 256, 0, stream>>>(encodings, IDX, EXP);

    // ---- phase 3: f0 predictor on expanded (T=2048) ----
    hipMemcpyAsync(X, EXP, (size_t)NBIG * 4, hipMemcpyDeviceToDevice, stream);
    run_vp(stream, f0_p, f0_pw, X, PPR, Y, Q, Kb, Vb, Ctx, PA, PB,
           T_DEC, out_f0, 0);

    // ---- phase 4: out = expanded + f0 embedding ----
    k_final<<<(unsigned)((NBIG + 255) / 256), 256, 0, stream>>>(
        EXP, f0s, f0_emb_w, f0_emb_b, out_main);
}
